// Model_88716844466342
// MI455X (gfx1250) — compile-verified, matrix operand order checked by reading, not run on hardware
//
#include <hip/hip_runtime.h>
#include <math.h>

// ---------------------------------------------------------------------------
// Types for CDNA5 WMMA
// ---------------------------------------------------------------------------
typedef _Float16 h16;
typedef __attribute__((ext_vector_type(8)))  h16   h8v;
typedef __attribute__((ext_vector_type(16))) h16   v16h;
typedef __attribute__((ext_vector_type(8)))  float v8f;

#define LN_EPS 1e-5f

__device__ __forceinline__ float siluf(float v) { return v / (1.0f + __expf(-v)); }

__device__ __forceinline__ float epilogue_act(float v, int act) {
    if (act == 1)      v = 0.5f * v * (1.0f + erff(v * 0.70710678118654752f));
    else if (act == 2) v = siluf(v);
    else if (act == 3) v = (v > 20.0f) ? v : log1pf(__expf(v));
    return v;
}

__device__ __forceinline__ v16h frag16(h8v lo, h8v hi) {
    return __builtin_shufflevector(lo, hi, 0,1,2,3,4,5,6,7,8,9,10,11,12,13,14,15);
}

// ---------------------------------------------------------------------------
// Simple WMMA GEMM (one wave per 16x16 tile, direct from global).
// Used for small / non-128-multiple N.  K%32==0, Npad%16==0, M%16==0.
// C[M,N] = act(A[M,K] @ W[N,K]^T + bias + addend)
// ---------------------------------------------------------------------------
__global__ void gemm_wmma_kernel(const h16* __restrict__ A, int lda,
                                 const h16* __restrict__ W, int ldw,
                                 float* C, int ldc,
                                 const float* __restrict__ bias,
                                 const float* addend, int ldadd,
                                 int K, int Nreal, int act)
{
    const int lane = threadIdx.x;       // 0..31, full wave
    const int r16  = lane & 15;
    const int hi   = lane >> 4;         // 0 or 1
    const int tn   = blockIdx.x;
    const int tm   = blockIdx.y;

    const h16* Arow = A + (size_t)(tm * 16 + r16) * lda;
    const h16* Wrow = W + (size_t)(tn * 16 + r16) * ldw;

    v8f acc = {};
    for (int kb = 0; kb < K; kb += 32) {
        const int k0 = kb + hi * 8;         // lanes<16: K 0..7   ; lanes>=16: K 8..15
        const int k1 = kb + 16 + hi * 8;    // lanes<16: K 16..23 ; lanes>=16: K 24..31
        h8v a0 = *(const h8v*)(Arow + k0);
        h8v a1 = *(const h8v*)(Arow + k1);
        h8v b0 = *(const h8v*)(Wrow + k0);
        h8v b1 = *(const h8v*)(Wrow + k1);
        if (kb + 32 < K) {                  // gfx1250 global_prefetch_b8
            __builtin_prefetch(Arow + k0 + 32, 0, 1);
            __builtin_prefetch(Wrow + k0 + 32, 0, 1);
        }
        acc = __builtin_amdgcn_wmma_f32_16x16x32_f16(
                  false, frag16(a0, a1), false, frag16(b0, b1),
                  (short)0, acc, false, false);
    }

    const int col = tn * 16 + r16;
    if (col >= Nreal) return;               // divergence only after WMMA loop
    const float bv_ = bias ? bias[col] : 0.0f;
#pragma unroll
    for (int r = 0; r < 8; ++r) {
        const int row = tm * 16 + r + hi * 8;   // C layout: vgpr r, lane-half selects M+8
        float v = acc[r] + bv_;
        if (addend) v += addend[(size_t)row * ldadd + col];
        C[(size_t)row * ldc + col] = epilogue_act(v, act);
    }
}

// ---------------------------------------------------------------------------
// LDS-tiled WMMA GEMM with double-buffered ASYNC staging:
// 256 threads (8 waves) per block, 128x128 C macro-tile, BK=32.
// Wave w computes rows [w*16, w*16+16) x all 128 cols: 8 accumulators, one A
// fragment reused against 8 B fragments per K step.  A/B tiles are DMA'd
// global->LDS with GLOBAL_LOAD_ASYNC_TO_LDS_B128 (ASYNCcnt), double buffered
// so the DMA of tile k+1 overlaps the 8-WMMA burst on tile k.
// Requires M%128==0, N%128==0, K%32==0.
// ---------------------------------------------------------------------------
#define TS   128
#define TPAD 40   // halves per LDS row (32 data + 8 pad) -> conflict-free b128

__global__ void gemm_wmma_tiled_kernel(const h16* __restrict__ A, int lda,
                                       const h16* __restrict__ W, int ldw,
                                       float* C, int ldc,
                                       const float* __restrict__ bias,
                                       const float* addend, int ldadd,
                                       int K, int act)
{
    __shared__ alignas(16) h16 lA[2][TS * TPAD];
    __shared__ alignas(16) h16 lB[2][TS * TPAD];

    const int tid  = threadIdx.x;        // 0..255
    const int wave = tid >> 5;           // 0..7
    const int lane = tid & 31;
    const int r16  = lane & 15;
    const int hi   = lane >> 4;

    const size_t rowA0 = (size_t)blockIdx.y * TS;   // M base
    const size_t rowW0 = (size_t)blockIdx.x * TS;   // N base

    // staging coordinates: thread covers chunk tid (rows 0..63) and tid+256
    // (rows 64..127); each chunk is 16 bytes (8 halves).
    const int srow0 = tid >> 2;
    const int srow1 = srow0 + 64;
    const int sch   = (tid & 3) * 8;

    // async DMA of one 32-wide K slice of A and B into LDS buffer `buf`
    auto stage = [&](int kb, int buf) {
        const h16* g0 = A + (rowA0 + srow0) * lda + kb + sch;
        const h16* g1 = A + (rowA0 + srow1) * lda + kb + sch;
        const h16* g2 = W + (rowW0 + srow0) * ldw + kb + sch;
        const h16* g3 = W + (rowW0 + srow1) * ldw + kb + sch;
        unsigned l0 = (unsigned)(size_t)&lA[buf][srow0 * TPAD + sch];
        unsigned l1 = (unsigned)(size_t)&lA[buf][srow1 * TPAD + sch];
        unsigned l2 = (unsigned)(size_t)&lB[buf][srow0 * TPAD + sch];
        unsigned l3 = (unsigned)(size_t)&lB[buf][srow1 * TPAD + sch];
        asm volatile("global_load_async_to_lds_b128 %0, %1, off" :: "v"(l0), "v"(g0) : "memory");
        asm volatile("global_load_async_to_lds_b128 %0, %1, off" :: "v"(l1), "v"(g1) : "memory");
        asm volatile("global_load_async_to_lds_b128 %0, %1, off" :: "v"(l2), "v"(g2) : "memory");
        asm volatile("global_load_async_to_lds_b128 %0, %1, off" :: "v"(l3), "v"(g3) : "memory");
    };

    v8f acc[8];
#pragma unroll
    for (int n = 0; n < 8; ++n) acc[n] = (v8f){};

    // prologue: fill buffer 0
    stage(0, 0);
    asm volatile("s_wait_asynccnt 0x0" ::: "memory");
    __syncthreads();

    int buf = 0;
    for (int kb = 0; kb < K; kb += 32) {
        const int nxt = buf ^ 1;
        if (kb + 32 < K) stage(kb + 32, nxt);   // DMA next tile during compute

        // ---- compute: one A fragment, 8 B fragments, 8 WMMAs
        const h16* pa = &lA[buf][(wave * 16 + r16) * TPAD];
        h8v a0 = *(const h8v*)(pa + hi * 8);
        h8v a1 = *(const h8v*)(pa + 16 + hi * 8);
        v16h av = frag16(a0, a1);
#pragma unroll
        for (int n = 0; n < 8; ++n) {
            const h16* pb = &lB[buf][(n * 16 + r16) * TPAD];
            h8v b0 = *(const h8v*)(pb + hi * 8);
            h8v b1 = *(const h8v*)(pb + 16 + hi * 8);
            acc[n] = __builtin_amdgcn_wmma_f32_16x16x32_f16(
                         false, av, false, frag16(b0, b1),
                         (short)0, acc[n], false, false);
        }

        // my DMA into `nxt` landed; barrier also proves all waves are done
        // reading `buf` before it is re-staged next iteration.
        asm volatile("s_wait_asynccnt 0x0" ::: "memory");
        __syncthreads();
        buf = nxt;
    }

    // ---- epilogue: wave writes its 16x128 strip
    const size_t rowBase = rowA0 + wave * 16;
#pragma unroll
    for (int n = 0; n < 8; ++n) {
        const size_t col = rowW0 + n * 16 + r16;
        const float bv_ = bias ? bias[col] : 0.0f;
#pragma unroll
        for (int r = 0; r < 8; ++r) {
            const size_t row = rowBase + r + hi * 8;
            float v = acc[n][r] + bv_;
            if (addend) v += addend[row * ldadd + col];
            C[row * ldc + col] = epilogue_act(v, act);
        }
    }
}

// ---------------------------------------------------------------------------
// f32 (strided, column-offset) -> f16 with zero padding of rows/cols
// ---------------------------------------------------------------------------
__global__ void convert_pad_kernel(const float* __restrict__ src, int srcLd,
                                   long srcRows, int srcCols, int colOff,
                                   h16* __restrict__ dst, int dstCols, long total)
{
    long i = (long)blockIdx.x * blockDim.x + threadIdx.x;
    if (i >= total) return;
    int  c = (int)(i % dstCols);
    long r = i / dstCols;
    float v = 0.0f;
    if (r < srcRows && c < srcCols) v = src[r * (long)srcLd + colOff + c];
    dst[i] = (h16)v;
}

// ---------------------------------------------------------------------------
// Causal depthwise conv1d (k=4) + SiLU.  xi lives in xz[:, 0:di] (ld = 2*di).
// ---------------------------------------------------------------------------
__global__ void conv_silu_kernel(const float* __restrict__ xz, int twodi,
                                 const float* __restrict__ cw, const float* __restrict__ cb,
                                 float* __restrict__ out, int L, int di, long total)
{
    long i = (long)blockIdx.x * blockDim.x + threadIdx.x;
    if (i >= total) return;
    int  c    = (int)(i % di);
    long rest = i / di;
    int  t    = (int)(rest % L);
    long seq  = rest / L;
    float s = cb[c];
#pragma unroll
    for (int k = 0; k < 4; ++k) {
        int tt = t - 3 + k;
        if (tt >= 0) s += xz[(seq * L + tt) * (long)twodi + c] * cw[c * 4 + k];
    }
    out[i] = siluf(s);
}

// ---------------------------------------------------------------------------
// Selective-scan. One thread per (seq, channel); h[16]/A[16] in registers.
// Reads dt[M,di], B/C from dbc[M,ld_dbc] at cols dtr / dtr+16, u=xi_conv[M,di],
// z from xz[:, di + c].  Writes y = (scan + u*D) * silu(z) into xz[:, c].
// ---------------------------------------------------------------------------
__global__ void ssm_scan_kernel(const float* __restrict__ dt,
                                const float* __restrict__ dbc, int ld_dbc, int dtr,
                                const float* __restrict__ u,
                                float* xz, int twodi,
                                const float* __restrict__ A_log,
                                const float* __restrict__ Dv,
                                int L, int di, long total)
{
    long i = (long)blockIdx.x * blockDim.x + threadIdx.x;
    if (i >= total) return;
    int  c   = (int)(i % di);
    long seq = i / di;

    float a[16], h[16];
#pragma unroll
    for (int j = 0; j < 16; ++j) {
        a[j] = -__expf(A_log[c * 16 + j]);
        h[j] = 0.0f;
    }
    const float dcoef = Dv[c];

    for (int t = 0; t < L; ++t) {
        const long  m   = seq * L + t;
        const float dtv = dt[m * (long)di + c];
        const float uv  = u[m * (long)di + c];
        const float du  = dtv * uv;
        const float* bc = dbc + m * (long)ld_dbc + dtr;
        float acc = 0.0f;
#pragma unroll
        for (int j = 0; j < 16; ++j) {
            float hj = h[j] * __expf(dtv * a[j]) + du * bc[j];
            h[j] = hj;
            acc += hj * bc[16 + j];
        }
        const float zv = xz[m * (long)twodi + di + c];
        xz[m * (long)twodi + c] = (acc + uv * dcoef) * siluf(zv);
    }
}

// ---------------------------------------------------------------------------
// LayerNorm over last dim D (in place, one thread per row)
// ---------------------------------------------------------------------------
__global__ void layernorm_kernel(float* __restrict__ x,
                                 const float* __restrict__ w, const float* __restrict__ b,
                                 int D, long rows)
{
    long r = (long)blockIdx.x * blockDim.x + threadIdx.x;
    if (r >= rows) return;
    float* p = x + r * (long)D;
    float m = 0.0f;
    for (int d = 0; d < D; ++d) m += p[d];
    m /= (float)D;
    float v = 0.0f;
    for (int d = 0; d < D; ++d) { float t = p[d] - m; v += t * t; }
    v /= (float)D;
    const float inv = rsqrtf(v + LN_EPS);
    for (int d = 0; d < D; ++d) p[d] = (p[d] - m) * inv * w[d] + b[d];
}

// ---------------------------------------------------------------------------
// RevIN statistics: x [4,512,32]; 128 (b,m) pairs
// ---------------------------------------------------------------------------
__global__ void revin_stats_kernel(const float* __restrict__ x,
                                   float* __restrict__ mean, float* __restrict__ stdv)
{
    int i = blockIdx.x * blockDim.x + threadIdx.x;
    if (i >= 128) return;
    int b = i >> 5, m = i & 31;
    const float* p = x + (long)b * 512 * 32 + m;
    float s = 0.0f;
    for (int l = 0; l < 512; ++l) s += p[l * 32];
    s /= 512.0f;
    float v = 0.0f;
    for (int l = 0; l < 512; ++l) { float t = p[l * 32] - s; v += t * t; }
    v /= 512.0f;
    mean[i] = s;
    stdv[i] = sqrtf(v + LN_EPS);
}

// ---------------------------------------------------------------------------
// RevIN normalize + start_fc embed: out[(bm*512+l)*64+d]
// ---------------------------------------------------------------------------
__global__ void revin_embed_kernel(const float* __restrict__ x,
                                   const float* __restrict__ mean, const float* __restrict__ stdv,
                                   const float* __restrict__ rw, const float* __restrict__ rb,
                                   const float* __restrict__ sw, const float* __restrict__ sb,
                                   float* __restrict__ out, long total)
{
    long i = (long)blockIdx.x * blockDim.x + threadIdx.x;
    if (i >= total) return;
    int  d    = (int)(i & 63);
    long rest = i >> 6;
    int  l    = (int)(rest % 512);
    int  bm   = (int)(rest / 512);
    int  b = bm >> 5, m = bm & 31;
    float xv = x[((long)b * 512 + l) * 32 + m];
    float xn = (xv - mean[bm]) / stdv[bm] * rw[m] + rb[m];
    out[i] = xn * sw[d] + sb[d];
}

// ---------------------------------------------------------------------------
// Head transpose + RevIN denorm: d_out[b, t, m], head [128, 96]
// ---------------------------------------------------------------------------
__global__ void head_denorm_kernel(const float* __restrict__ head,
                                   const float* __restrict__ mean, const float* __restrict__ stdv,
                                   const float* __restrict__ rw, const float* __restrict__ rb,
                                   float* __restrict__ out)
{
    int i = blockIdx.x * blockDim.x + threadIdx.x;
    if (i >= 4 * 96 * 32) return;
    int m = i & 31;
    int rest = i >> 5;
    int t = rest % 96;
    int b = rest / 96;
    int bm = b * 32 + m;
    float v = head[(long)bm * 96 + t];
    out[i] = (v - rb[m]) / (rw[m] + 1e-10f) * stdv[bm] + mean[bm];
}

// ---------------------------------------------------------------------------
// Host orchestration
// ---------------------------------------------------------------------------
extern "C" void kernel_launch(void* const* d_in, const int* in_sizes, int n_in,
                              void* d_out, int out_size, void* d_ws, size_t ws_size,
                              hipStream_t stream)
{
    (void)in_sizes; (void)n_in; (void)out_size; (void)ws_size;

    auto in = [&](int i) -> const float* { return (const float*)d_in[i]; };

    // ---- workspace carve-out (256B aligned) ----
    char*  base = (char*)d_ws;
    size_t off  = 0;
    auto alloc = [&](size_t bytes) -> void* {
        off = (off + 255) & ~(size_t)255;
        void* p = base + off;
        off += bytes;
        return p;
    };
    float* f_x    = (float*)alloc((size_t)65536 * 64 * 4);        //  16.8 MB activations [128,512,64]
    float* f_xz   = (float*)alloc((size_t)16777216 * 4);          //  67 MB  xz (max 4096x4096 / 65536x256)
    float* f_conv = (float*)alloc((size_t)8388608 * 4);           //  33 MB  xi_conv / FFN hidden
    float* f_dbc  = (float*)alloc((size_t)65536 * 48 * 4);        //  12.6MB dbc (ld padded)
    float* f_dt   = (float*)alloc((size_t)8388608 * 4);           //  33 MB  dt
    float* f_mean = (float*)alloc(128 * 4);
    float* f_std  = (float*)alloc(128 * 4);
    float* f_head = (float*)alloc((size_t)128 * 96 * 4);
    h16*   h_A    = (h16*)alloc((size_t)8388608 * 2);             //  16.8MB f16 A staging
    h16*   h_W    = (h16*)alloc((size_t)4194304 * 2);             //   8.4MB f16 W staging

    // ---- launch helpers ----
    auto convert = [&](const float* src, int srcLd, long srcRows, int srcCols, int colOff,
                       h16* dst, long dstRows, int dstCols) {
        long total  = dstRows * (long)dstCols;
        long blocks = (total + 255) / 256;
        convert_pad_kernel<<<dim3((unsigned)blocks), 256, 0, stream>>>(
            src, srcLd, srcRows, srcCols, colOff, dst, dstCols, total);
    };
    auto gemm = [&](const h16* A, int lda, const h16* W, int ldw, float* C, int ldc,
                    const float* bias, const float* addend, int ldadd,
                    int M, int Npad, int Nreal, int K, int act) {
        if ((Npad % 128 == 0) && (M % 128 == 0) && (Nreal == Npad)) {
            dim3 grid((unsigned)(Npad / 128), (unsigned)(M / 128));
            gemm_wmma_tiled_kernel<<<grid, 256, 0, stream>>>(
                A, lda, W, ldw, C, ldc, bias, addend, ldadd, K, act);
        } else {
            dim3 grid((unsigned)(Npad / 16), (unsigned)(M / 16));
            gemm_wmma_kernel<<<grid, 32, 0, stream>>>(
                A, lda, W, ldw, C, ldc, bias, addend, ldadd, K, Nreal, act);
        }
    };
    auto ln = [&](float* x, const float* w, const float* b, long rows, int D) {
        long blocks = (rows + 255) / 256;
        layernorm_kernel<<<dim3((unsigned)blocks), 256, 0, stream>>>(x, w, b, D, rows);
    };

    // ---- one mamba_ssm block (in place: x <- x + mamba(x)) ----
    auto run_mamba = [&](float* x, int Nseq, int L, int d, int di, int dtr, int dtrPad,
                         const float* in_proj, const float* conv_w, const float* conv_b,
                         const float* x_proj, const float* dt_w, const float* dt_b,
                         const float* A_log, const float* Dv, const float* out_proj) {
        const long M      = (long)Nseq * L;
        const int  twodi  = 2 * di;
        const int  nbc    = dtr + 32;              // dtr + 2*n (n=16)
        const int  nbcPad = (nbc + 15) & ~15;

        // xz = x @ in_proj.T                                  [M, 2di]
        convert(x, d, M, d, 0, h_A, M, d);
        convert(in_proj, d, twodi, d, 0, h_W, twodi, d);
        gemm(h_A, d, h_W, d, f_xz, twodi, nullptr, nullptr, 0,
             (int)M, twodi, twodi, d, 0);

        // xi = silu(causal_conv(xz[:, :di]))                  [M, di]
        {
            long total = M * di;
            conv_silu_kernel<<<dim3((unsigned)((total + 255) / 256)), 256, 0, stream>>>(
                f_xz, twodi, conv_w, conv_b, f_conv, L, di, total);
        }

        // dbc = xi @ x_proj.T                                 [M, dtr+32]
        convert(f_conv, di, M, di, 0, h_A, M, di);
        convert(x_proj, di, nbc, di, 0, h_W, nbcPad, di);
        gemm(h_A, di, h_W, di, f_dbc, nbcPad, nullptr, nullptr, 0,
             (int)M, nbcPad, nbc, di, 0);

        // dt = softplus(dbc[:, :dtr] @ dt_w.T + dt_b)         [M, di]
        convert(f_dbc, nbcPad, M, dtr, 0, h_A, M, dtrPad);
        convert(dt_w, dtr, di, dtr, 0, h_W, di, dtrPad);
        gemm(h_A, dtrPad, h_W, dtrPad, f_dt, di, dt_b, nullptr, 0,
             (int)M, di, di, dtrPad, 3);

        // selective scan; y written into f_xz[:, :di]
        {
            long total = (long)Nseq * di;
            ssm_scan_kernel<<<dim3((unsigned)((total + 255) / 256)), 256, 0, stream>>>(
                f_dt, f_dbc, nbcPad, dtr, f_conv, f_xz, twodi, A_log, Dv, L, di, total);
        }

        // x += y @ out_proj.T  (residual fused into epilogue)
        convert(f_xz, twodi, M, di, 0, h_A, M, di);
        convert(out_proj, di, d, di, 0, h_W, d, di);
        gemm(h_A, di, h_W, di, x, d, nullptr, x, d,
             (int)M, d, d, di, 0);
    };

    // ---- FFN: x += GELU(x@w1.T + b1) @ w2.T + b2 ----
    auto run_ffn = [&](float* x, const float* w1, const float* b1,
                       const float* w2, const float* b2) {
        const long M = 65536;
        convert(x, 64, M, 64, 0, h_A, M, 64);
        convert(w1, 64, 128, 64, 0, h_W, 128, 64);
        gemm(h_A, 64, h_W, 64, f_conv, 128, b1, nullptr, 0, (int)M, 128, 128, 64, 1);
        convert(f_conv, 128, M, 128, 0, h_A, M, 128);
        convert(w2, 128, 64, 128, 0, h_W, 64, 128);
        gemm(h_A, 128, h_W, 128, x, 64, b2, x, 64, (int)M, 64, 64, 128, 0);
    };

    // ================= forward =================
    const float* x_in = in(0);

    // RevIN stats + normalize + start_fc -> f_x [128, 512, 64]
    revin_stats_kernel<<<4, 32, 0, stream>>>(x_in, f_mean, f_std);
    {
        long total = (long)128 * 512 * 64;
        revin_embed_kernel<<<dim3((unsigned)((total + 255) / 256)), 256, 0, stream>>>(
            x_in, f_mean, f_std, in(1), in(2), in(3), in(4), f_x, total);
    }

    // encoder layers (params start at index 5, 26 leaves per layer)
    for (int li = 0; li < 2; ++li) {
        const int p = 5 + li * 26;
        // m1: seq mamba on [128, 512, 64]
        run_mamba(f_x, 128, 512, 64, 128, 4, 32,
                  in(p + 0), in(p + 1), in(p + 2), in(p + 3), in(p + 4),
                  in(p + 5), in(p + 6), in(p + 7), in(p + 8));
        ln(f_x, in(p + 18), in(p + 19), 65536, 64);
        // m2: patch mamba on contiguous reshape [128, 32, 1024]
        run_mamba(f_x, 128, 32, 1024, 2048, 64, 64,
                  in(p + 9), in(p + 10), in(p + 11), in(p + 12), in(p + 13),
                  in(p + 14), in(p + 15), in(p + 16), in(p + 17));
        // FFN + norm2
        run_ffn(f_x, in(p + 20), in(p + 21), in(p + 22), in(p + 23));
        ln(f_x, in(p + 24), in(p + 25), 65536, 64);
    }

    // final norm
    ln(f_x, in(57), in(58), 65536, 64);

    // projection head: [128, 32768] @ proj_w[96, 32768].T + proj_b
    convert(f_x, 32768, 128, 32768, 0, h_A, 128, 32768);
    convert(in(59), 32768, 96, 32768, 0, h_W, 96, 32768);
    gemm(h_A, 32768, h_W, 32768, f_head, 96, in(60), nullptr, 0,
         128, 96, 96, 32768, 0);

    // transpose + RevIN denormalize -> d_out [4, 96, 32]
    head_denorm_kernel<<<48, 256, 0, stream>>>(f_head, f_mean, f_std,
                                               in(1), in(2), (float*)d_out);
}
